// GINe_8160437862721
// MI455X (gfx1250) — compile-verified, matrix operand order checked by reading, not run on hardware
//
#include <hip/hip_runtime.h>

// ---------------------------------------------------------------------------
// GINE GNN forward for MI455X (gfx1250, wave32, WMMA + TDM).
// - All GEMMs: v_wmma_f32_16x16x32_f16, f32 accumulate.
// - Edge-sized intermediates stored f16 (halves dominant HBM traffic).
// - Dominant (f16-input) GEMMs: BOTH A and B tiles DMA'd by the Tensor Data
//   Mover, double-buffered in LDS for DMA/compute overlap.
// - f32-input / gather GEMMs: VALU-staged A, TDM-staged B.
// ---------------------------------------------------------------------------

typedef __attribute__((ext_vector_type(16))) _Float16 v16h;
typedef __attribute__((ext_vector_type(8)))  _Float16 v8h;
typedef __attribute__((ext_vector_type(8)))  float    v8f;
typedef __attribute__((ext_vector_type(4)))  unsigned int v4u;
typedef __attribute__((ext_vector_type(8)))  int v8i;
typedef __attribute__((ext_vector_type(4)))  int v4i;

#define GN 50000
#define GE 800000
#define GF 128
#define GED 16
#define GH 100

// ---------------------------------------------------------------------------
// TDM: DMA a (tile1 x tile0) f16 tile whose origin is gptr into LDS (row-major,
// tile0 halfs per row). rem0/rem1 = remaining tensor extent from the tile
// origin (hardware OOB zero-fills beyond). stride0 = row stride in elements.
// Descriptor packing per CDNA5 ISA §8.3/§8.4 (2D tensor, data_size=2B).
// ---------------------------------------------------------------------------
__device__ inline void tdm_load_2d(const _Float16* gptr, int rem0, int rem1,
                                   int stride0, int tile0, int tile1, void* ldsdst)
{
    unsigned long long gaddr = (unsigned long long)(const void*)gptr;
    unsigned int lds_addr = (unsigned int)(unsigned long long)(size_t)ldsdst;

    // Group 0: count=1 | lds_addr<<32 ; global_addr[56:0] | type=2 at [127:126]
    unsigned long long qa = 1ull | ((unsigned long long)lds_addr << 32);
    unsigned long long qb = (gaddr & ((1ull << 57) - 1)) | (2ull << 62);
    v4u g0 = { (unsigned)qa, (unsigned)(qa >> 32),
               (unsigned)qb, (unsigned)(qb >> 32) };

    // Group 1 (256b): wg_mask=0, data_size=1(2B) at [17:16],
    //   tensor_dim0 at [79:48], tensor_dim1 at [111:80],
    //   tile_dim0 at [127:112], tile_dim1 at [143:128],
    //   tensor_dim0_stride at [207:160], tensor_dim1_stride=0.
    unsigned long long D0 = (unsigned long long)(unsigned)rem0;
    unsigned long long S0 = (unsigned long long)(unsigned)stride0;
    unsigned long long q0 = 0x10000ull | ((D0 & 0xFFFFull) << 48);
    unsigned long long q1 = ((D0 >> 16) & 0xFFFFull)
                          | ((unsigned long long)(unsigned)rem1 << 16)
                          | ((unsigned long long)(unsigned)tile0 << 48);
    unsigned long long q2 = (unsigned long long)(unsigned)tile1
                          | ((S0 & 0xFFFFFFFFull) << 32);
    unsigned long long q3 = (S0 >> 32) & 0xFFFFull;
    v8i g1 = { (int)q0, (int)(q0 >> 32), (int)q1, (int)(q1 >> 32),
               (int)q2, (int)(q2 >> 32), (int)q3, (int)(q3 >> 32) };
    v4i gz = { 0, 0, 0, 0 };
#if __has_include(<hip/amd_detail/amd_gfx1250_TDM.h>)
    v8i gz8 = { 0, 0, 0, 0, 0, 0, 0, 0 };
    __builtin_amdgcn_tensor_load_to_lds(g0, g1, gz, gz, gz8, 0);
#else
    __builtin_amdgcn_tensor_load_to_lds(g0, g1, gz, gz, 0);
#endif
}

// Transpose + f32->f16 convert a weight matrix: Wt[n*K+k] = W[k*Nout+n].
__global__ __launch_bounds__(256)
void wprep_kernel(const float* __restrict__ W, _Float16* __restrict__ Wt,
                  int K, int Nout)
{
    int i = blockIdx.x * 256 + threadIdx.x;
    if (i < K * Nout) {
        int k = i / Nout, n = i % Nout;
        Wt[(size_t)n * K + k] = (_Float16)W[i];
    }
}

// Compute one 128 x NT*16 block-tile worth of WMMAs from staged LDS tiles.
template<int NT>
__device__ inline void wmma_tiles(const _Float16* sA, const _Float16* sB,
                                  int wave, int lane, v8f (&acc)[NT])
{
    const int r  = lane & 15;
    const int kb = (lane < 16) ? 0 : 8;
    const int abase = (wave * 16 + r) * 32 + kb;
    v16h afrag;
    ((v8h*)&afrag)[0] = *(const v8h*)&sA[abase];
    ((v8h*)&afrag)[1] = *(const v8h*)&sA[abase + 16];
#pragma unroll
    for (int t = 0; t < NT; ++t) {
        const int nbase = (t * 16 + r) * 32 + kb;
        v16h bfrag;
        ((v8h*)&bfrag)[0] = *(const v8h*)&sB[nbase];
        ((v8h*)&bfrag)[1] = *(const v8h*)&sB[nbase + 16];
        acc[t] = __builtin_amdgcn_wmma_f32_16x16x32_f16(
            false, afrag, false, bfrag, (short)0, acc[t], false, false);
    }
}

template<int NT, bool RELU, typename TOUT>
__device__ inline void gemm_epilogue(v8f (&acc)[NT], const float* bias,
                                     TOUT* C, int row0, int wave, int lane,
                                     int M, int Nout)
{
    const int mrow0 = row0 + wave * 16 + ((lane < 16) ? 0 : 8);
#pragma unroll
    for (int t = 0; t < NT; ++t) {
        int n = t * 16 + (lane & 15);
        if (n >= Nout) continue;
        float bv = bias ? bias[n] : 0.f;
#pragma unroll
        for (int i = 0; i < 8; ++i) {
            int m = mrow0 + i;
            if (m < M) {
                float v = acc[t][i] + bv;
                if (RELU) v = fmaxf(v, 0.f);
                C[(size_t)m * Nout + n] = (TOUT)v;
            }
        }
    }
}

// ---------------------------------------------------------------------------
// Fully-TDM, double-buffered WMMA GEMM for f16 inputs:
//   C[M,Nout] = act(A[M,K](f16) @ W + bias),  W given pre-transposed f16.
// Wave 0 issues the next chunk's A+B DMAs before computing the current chunk.
// ---------------------------------------------------------------------------
template<int NT, bool RELU, typename TOUT>
__global__ __launch_bounds__(256)
void gemm_f16_kernel(const _Float16* __restrict__ A,
                     const _Float16* __restrict__ Wt,
                     const float* __restrict__ bias, TOUT* __restrict__ C,
                     int M, int K, int Nout)
{
    __shared__ __align__(16) _Float16 sA[2][128 * 32];
    __shared__ __align__(16) _Float16 sB[2][NT * 16 * 32];

    const int tid  = threadIdx.x;
    const int wave = tid >> 5;
    const int lane = tid & 31;
    const int row0 = blockIdx.x * 128;
    const int rows_rem = M - row0;

    v8f acc[NT];
    const v8f vzero = {0.f, 0.f, 0.f, 0.f, 0.f, 0.f, 0.f, 0.f};
#pragma unroll
    for (int t = 0; t < NT; ++t) acc[t] = vzero;

    const int nk = (K + 31) / 32;
    // Prologue: DMA chunk 0.
    if (tid < 32) {
        tdm_load_2d(A + (size_t)row0 * K, K, rows_rem, K, 32, 128, sA[0]);
        tdm_load_2d(Wt, K, Nout, K, 32, NT * 16, sB[0]);
        __builtin_amdgcn_s_wait_tensorcnt(0);
    }
    __syncthreads();

    for (int kc = 0; kc < nk; ++kc) {
        const int cur = kc & 1;
        const int nxt = cur ^ 1;
        const bool more = (kc + 1) < nk;
        if (more && tid < 32) {
            const int k0n = (kc + 1) * 32;
            tdm_load_2d(A + (size_t)row0 * K + k0n, K - k0n, rows_rem,
                        K, 32, 128, sA[nxt]);
            tdm_load_2d(Wt + k0n, K - k0n, Nout, K, 32, NT * 16, sB[nxt]);
        }
        wmma_tiles<NT>(sA[cur], sB[cur], wave, lane, acc);
        if (more && tid < 32) __builtin_amdgcn_s_wait_tensorcnt(0);
        __syncthreads();
    }

    gemm_epilogue<NT, RELU, TOUT>(acc, bias, C, row0, wave, lane, M, Nout);
}

// ---------------------------------------------------------------------------
// WMMA GEMM with VALU-staged (f32->f16) A, TDM-staged B.
// ---------------------------------------------------------------------------
template<int NT, bool RELU, typename TOUT>
__global__ __launch_bounds__(256)
void gemm_kernel(const float* __restrict__ A, const _Float16* __restrict__ Wt,
                 const float* __restrict__ bias, TOUT* __restrict__ C,
                 int M, int K, int Nout)
{
    __shared__ __align__(16) _Float16 sA[128 * 32];
    __shared__ __align__(16) _Float16 sB[NT * 16 * 32];

    const int tid  = threadIdx.x;
    const int wave = tid >> 5;
    const int lane = tid & 31;
    const int row0 = blockIdx.x * 128;

    v8f acc[NT];
    const v8f vzero = {0.f, 0.f, 0.f, 0.f, 0.f, 0.f, 0.f, 0.f};
#pragma unroll
    for (int t = 0; t < NT; ++t) acc[t] = vzero;

    for (int k0 = 0; k0 < K; k0 += 32) {
        if (tid < 32) tdm_load_2d(Wt + k0, K - k0, Nout, K, 32, NT * 16, sB);

        if (k0 + 32 < K)
            __builtin_prefetch(&A[(size_t)(row0 + (tid >> 3)) * K + k0 + 32], 0, 1);

#pragma unroll
        for (int i = tid; i < 128 * 32; i += 256) {
            int r = i >> 5, c = i & 31;
            int gr = row0 + r, gc = k0 + c;
            float v = (gr < M && gc < K) ? A[(size_t)gr * K + gc] : 0.f;
            sA[r * 32 + c] = (_Float16)v;
        }
        if (tid < 32) __builtin_amdgcn_s_wait_tensorcnt(0);
        __syncthreads();

        wmma_tiles<NT>(sA, sB, wave, lane, acc);
        __syncthreads();
    }

    gemm_epilogue<NT, RELU, TOUT>(acc, bias, C, row0, wave, lane, M, Nout);
}

// ---------------------------------------------------------------------------
// WMMA GEMM over a fused gather-concat A (K = 300):
//   A[e] = [ act01(X0[idx0[e]]), act01(X1[idx1[e]]), X2[e] ]
// X0/X1 node features (f32), X2 edge features (f16). TDM-staged B.
// ---------------------------------------------------------------------------
template<int NT, bool RELU_OUT, bool RELU01, typename TOUT>
__global__ __launch_bounds__(256)
void gemm_cat3_kernel(const float* __restrict__ X0, const int* __restrict__ idx0,
                      const float* __restrict__ X1, const int* __restrict__ idx1,
                      const _Float16* __restrict__ X2,
                      const _Float16* __restrict__ Wt, const float* __restrict__ bias,
                      TOUT* __restrict__ C, int M, int Nout)
{
    const int K = 3 * GH;  // 300
    __shared__ __align__(16) _Float16 sA[128 * 32];
    __shared__ __align__(16) _Float16 sB[NT * 16 * 32];

    const int tid  = threadIdx.x;
    const int wave = tid >> 5;
    const int lane = tid & 31;
    const int row0 = blockIdx.x * 128;

    v8f acc[NT];
    const v8f vzero = {0.f, 0.f, 0.f, 0.f, 0.f, 0.f, 0.f, 0.f};
#pragma unroll
    for (int t = 0; t < NT; ++t) acc[t] = vzero;

    for (int k0 = 0; k0 < K; k0 += 32) {
        if (tid < 32) tdm_load_2d(Wt + k0, K - k0, Nout, K, 32, NT * 16, sB);

#pragma unroll
        for (int i = tid; i < 128 * 32; i += 256) {
            int r = i >> 5, c = i & 31;
            int gr = row0 + r, gk = k0 + c;
            float v = 0.f;
            if (gr < M && gk < K) {
                if (gk < GH) {
                    v = X0[(size_t)idx0[gr] * GH + gk];
                    if (RELU01) v = fmaxf(v, 0.f);
                } else if (gk < 2 * GH) {
                    v = X1[(size_t)idx1[gr] * GH + (gk - GH)];
                    if (RELU01) v = fmaxf(v, 0.f);
                } else {
                    v = (float)X2[(size_t)gr * GH + (gk - 2 * GH)];
                }
            }
            sA[r * 32 + c] = (_Float16)v;
        }
        if (tid < 32) __builtin_amdgcn_s_wait_tensorcnt(0);
        __syncthreads();

        wmma_tiles<NT>(sA, sB, wave, lane, acc);
        __syncthreads();
    }

    gemm_epilogue<NT, RELU_OUT, TOUT>(acc, bias, C, row0, wave, lane, M, Nout);
}

// ---------------------------------------------------------------------------
// Edge message + scatter: agg[d[e]] += relu(h[s[e]] + tlin[e])
// ---------------------------------------------------------------------------
__global__ __launch_bounds__(256)
void edge_msg_kernel(const float* __restrict__ h, const _Float16* __restrict__ tlin,
                     const int* __restrict__ s, const int* __restrict__ d,
                     float* __restrict__ agg, int E)
{
    long long gid = (long long)blockIdx.x * 256 + threadIdx.x;
    int f = (int)(gid & 127);
    long long e = gid >> 7;
    if (e >= E || f >= GH) return;
    int si = s[e], di = d[e];
    float v = (float)tlin[(size_t)e * GH + f] + h[(size_t)si * GH + f];
    v = fmaxf(v, 0.f);
    unsafeAtomicAdd(&agg[(size_t)di * GH + f], v);
}

__global__ __launch_bounds__(256)
void add2_kernel(const float* __restrict__ a, const float* __restrict__ b,
                 float* __restrict__ o, long long n)
{
    long long i = (long long)blockIdx.x * 256 + threadIdx.x;
    if (i < n) o[i] = a[i] + b[i];
}

template<typename T>
__global__ __launch_bounds__(256)
void avg3_kernel(const T* __restrict__ a, const T* __restrict__ b,
                 const T* __restrict__ c, T* __restrict__ o, long long n)
{
    long long i = (long long)blockIdx.x * 256 + threadIdx.x;
    if (i < n)
        o[i] = (T)(((float)a[i] + (float)b[i] + (float)c[i]) * (1.0f / 3.0f));
}

__global__ __launch_bounds__(256)
void bn_stats_kernel(const float* __restrict__ z, float* __restrict__ stats, int n)
{
    long long gid = (long long)blockIdx.x * 256 + threadIdx.x;
    int f = (int)(gid & 127);
    long long r = gid >> 7;
    if (r >= n || f >= GH) return;
    float v = z[(size_t)r * GH + f];
    unsafeAtomicAdd(&stats[f], v);
    unsafeAtomicAdd(&stats[128 + f], v * v);
}

__global__ __launch_bounds__(256)
void bn_apply_kernel(const float* __restrict__ z, const float* __restrict__ stats,
                     const float* __restrict__ g, const float* __restrict__ b,
                     float* __restrict__ o, int n)
{
    long long gid = (long long)blockIdx.x * 256 + threadIdx.x;
    int f = (int)(gid & 127);
    long long r = gid >> 7;
    if (r >= n || f >= GH) return;
    float inv_n = 1.0f / (float)n;
    float mu  = stats[f] * inv_n;
    float var = stats[128 + f] * inv_n - mu * mu;
    float v = (z[(size_t)r * GH + f] - mu) * rsqrtf(var + 1e-5f) * g[f] + b[f];
    o[(size_t)r * GH + f] = fmaxf(v, 0.f);
}

// ---------------------------------------------------------------------------
// Orchestration
// ---------------------------------------------------------------------------
extern "C" void kernel_launch(void* const* d_in, const int* in_sizes, int n_in,
                              void* d_out, int out_size, void* d_ws, size_t ws_size,
                              hipStream_t stream)
{
    const int N = GN, E = GE, H = GH;
    const float* x     = (const float*)d_in[0];
    const float* eattr = (const float*)d_in[1];
    const int*   ei    = (const int*)d_in[2];
    const int* src = ei;
    const int* dst = ei + E;
    auto P = [&](int i) { return (const float*)d_in[i]; };
    // 3 nodeW, 4 nodeb, 5 edgeW, 6 edgeb; layer l base 7+l*24:
    //   f:+0..7, r:+8..15, ef:+16..19, er:+20..23; head base 55.

    char* wsb = (char*)d_ws;
    size_t off = 0;
    auto walloc = [&](size_t bytes) {
        void* r = wsb + off;
        off = (off + bytes + 255) & ~(size_t)255;
        return r;
    };
    float*     h     = (float*)    walloc((size_t)N * H * 4);
    _Float16*  ea    = (_Float16*) walloc((size_t)E * H * 2);
    _Float16*  b1    = (_Float16*) walloc((size_t)E * H * 2);
    _Float16*  b2    = (_Float16*) walloc((size_t)E * H * 2);
    _Float16*  b3    = (_Float16*) walloc((size_t)E * H * 2);
    float*     agg   = (float*)    walloc((size_t)N * H * 4);
    float*     tmp   = (float*)    walloc((size_t)N * H * 4);
    float*     u     = (float*)    walloc((size_t)N * H * 4);
    float*     zbuf  = (float*)    walloc((size_t)N * H * 4);
    float*     hf    = (float*)    walloc((size_t)N * H * 4);
    float*     hr    = (float*)    walloc((size_t)N * H * 4);
    float*     stats = (float*)    walloc(256 * 4);
    _Float16*  wt    = (_Float16*) walloc((size_t)300 * 112 * 2);

    const int gN  = (N + 127) / 128;
    const int gE  = (E + 127) / 128;
    const int gNH = (int)(((long long)N * 128 + 255) / 256);
    const int gEH = (int)(((long long)E * 128 + 255) / 256);
    const int eNH = (int)(((long long)N * H + 255) / 256);
    const int eEH = (int)(((long long)E * H + 255) / 256);

    auto prep = [&](int widx, int K, int Nout) {
        int n = K * Nout;
        wprep_kernel<<<(n + 255) / 256, 256, 0, stream>>>(P(widx), wt, K, Nout);
    };

    // Embeddings.
    prep(3, GF, H);
    gemm_kernel<7, false, float><<<gN, 256, 0, stream>>>(x, wt, P(4), h, N, GF, H);
    prep(5, GED, H);
    gemm_kernel<7, false, _Float16><<<gE, 256, 0, stream>>>(
        eattr, wt, P(6), ea, E, GED, H);

    auto gine_branch = [&](int pb, const int* s, const int* d, float* hout) {
        prep(pb + 0, H, H);                                              // lin
        gemm_f16_kernel<7, false, _Float16><<<gE, 256, 0, stream>>>(
            ea, wt, P(pb + 1), b1, E, H, H);
        hipMemsetAsync(agg, 0, (size_t)N * H * sizeof(float), stream);
        edge_msg_kernel<<<gEH, 256, 0, stream>>>(h, b1, s, d, agg, E);
        add2_kernel<<<eNH, 256, 0, stream>>>(h, agg, tmp, (long long)N * H);
        prep(pb + 2, H, H);                                              // mlp1
        gemm_kernel<7, true, float><<<gN, 256, 0, stream>>>(
            tmp, wt, P(pb + 3), u, N, H, H);
        prep(pb + 4, H, H);                                              // mlp2
        gemm_kernel<7, false, float><<<gN, 256, 0, stream>>>(
            u, wt, P(pb + 5), zbuf, N, H, H);
        hipMemsetAsync(stats, 0, 256 * sizeof(float), stream);
        bn_stats_kernel<<<gNH, 256, 0, stream>>>(zbuf, stats, N);
        bn_apply_kernel<<<gNH, 256, 0, stream>>>(zbuf, stats, P(pb + 6),
                                                 P(pb + 7), hout, N);
    };

    for (int l = 0; l < 2; ++l) {
        const int L = 7 + l * 24;
        gine_branch(L + 0, src, dst, hf);
        gine_branch(L + 8, dst, src, hr);
        avg3_kernel<float><<<eNH, 256, 0, stream>>>(h, hf, hr, h,
                                                    (long long)N * H);

        prep(L + 16, 3 * H, H);                                          // ef.l1
        gemm_cat3_kernel<7, true, false, _Float16><<<gE, 256, 0, stream>>>(
            h, src, h, dst, ea, wt, P(L + 17), b3, E, H);
        prep(L + 18, H, H);                                              // ef.l2
        gemm_f16_kernel<7, false, _Float16><<<gE, 256, 0, stream>>>(
            b3, wt, P(L + 19), b1, E, H, H);
        prep(L + 20, 3 * H, H);                                          // er.l1
        gemm_cat3_kernel<7, true, false, _Float16><<<gE, 256, 0, stream>>>(
            h, dst, h, src, ea, wt, P(L + 21), b3, E, H);
        prep(L + 22, H, H);                                              // er.l2
        gemm_f16_kernel<7, false, _Float16><<<gE, 256, 0, stream>>>(
            b3, wt, P(L + 23), b2, E, H, H);
        avg3_kernel<_Float16><<<eEH, 256, 0, stream>>>(ea, b1, b2, ea,
                                                       (long long)E * H);
    }

    // Head: cat(relu(h[src]), relu(h[dst]), ea) -> 50 -> 25 -> 2.
    prep(55, 3 * H, 50);
    gemm_cat3_kernel<4, true, true, _Float16><<<gE, 256, 0, stream>>>(
        h, src, h, dst, ea, wt, P(56), b3, E, 50);
    prep(57, 50, 25);
    gemm_f16_kernel<2, true, _Float16><<<gE, 256, 0, stream>>>(
        b3, wt, P(58), b1, E, 50, 25);
    prep(59, 25, 2);
    gemm_f16_kernel<1, false, float><<<gE, 256, 0, stream>>>(
        b1, wt, P(60), (float*)d_out, E, 25, 2);
}
